// new_knn_43052752175170
// MI455X (gfx1250) — matrix operand order repaired and checked by reading, not compiled, not run on hardware
//
#include <hip/hip_runtime.h>
#include <hip/hip_bf16.h>

// ---------------- problem constants ----------------
#define BB   16
#define CC   3
#define NN   2048          // both point count and feature dim
#define KT   50            // top-k candidates
#define KK   20            // final k
#define MM   150           // KT*CC
#define MP   160           // MM padded to multiple of 16
#define NG   4             // n-tiles (of 16) per wave in the GEMM => 64 columns/block

typedef __attribute__((ext_vector_type(16))) __bf16 v16bf;
typedef __attribute__((ext_vector_type(8)))  float  v8f;

union FragU { v16bf v; unsigned int u[8]; };

// CDNA5 async global->LDS path (ASYNCcnt), with sync fallback
#if defined(__has_builtin)
#if __has_builtin(__builtin_amdgcn_global_load_async_to_lds_b128) && \
    __has_builtin(__builtin_amdgcn_s_wait_asynccnt)
#define USE_ASYNC_LDS 1
#endif
#endif
#ifndef USE_ASYNC_LDS
#define USE_ASYNC_LDS 0
#endif

#if USE_ASYNC_LDS
// builtin parameter types (from compiler diagnostic): pointer to 16B int vector,
// param0 in global (AS1), param1 in LDS (AS3)
typedef int vint4 __attribute__((vector_size(4 * sizeof(int))));
typedef __attribute__((address_space(1))) vint4 g_vint4;
typedef __attribute__((address_space(3))) vint4 l_vint4;
#endif

__device__ __forceinline__ void stage16(unsigned short* lds_dst,
                                        const unsigned short* gsrc) {
#if USE_ASYNC_LDS
    __builtin_amdgcn_global_load_async_to_lds_b128(
        (g_vint4*)gsrc, (l_vint4*)lds_dst, 0, 0);
#else
    *reinterpret_cast<uint4*>(lds_dst) = *reinterpret_cast<const uint4*>(gsrc);
#endif
}

__device__ __forceinline__ void async_wait_all() {
#if USE_ASYNC_LDS
    __builtin_amdgcn_s_wait_asynccnt(0);
#endif
}

__device__ __forceinline__ unsigned short f2bf(float f) {
    union { float f; unsigned int u; } cv; cv.f = f;
    unsigned int u = cv.u;
    u += 0x7FFFu + ((u >> 16) & 1u);   // round-to-nearest-even
    return (unsigned short)(u >> 16);
}

// ---------------- 1) weight fp32 (N,K) -> bf16 transposed (K,N) ----------------
__global__ void k_w2bfT(const float* __restrict__ W, unsigned short* __restrict__ Wt,
                        int Nrow, int Kcol) {
    size_t t = (size_t)blockIdx.x * blockDim.x + threadIdx.x;
    if (t >= (size_t)Nrow * Kcol) return;
    int    n = (int)(t % Nrow);
    size_t k = t / Nrow;
    Wt[t] = f2bf(W[(size_t)n * Kcol + k]);
}

// ---------------- 2) KNN: top-50 by -|xi-xj|^2, one block per (b,n) ----------------
__global__ void k_knn(const float* __restrict__ x, int* __restrict__ idx) {
    __shared__ float dist[NN];
    __shared__ float rv[256];
    __shared__ int   ri[256];
    const int b = blockIdx.x / NN;
    const int n = blockIdx.x % NN;
    const int tid = threadIdx.x;
    const float* xb = x + (size_t)b * CC * NN;
    const float q0 = xb[n], q1 = xb[NN + n], q2 = xb[2 * NN + n];
    for (int m = tid; m < NN; m += 256) {
        float d0 = q0 - xb[m], d1 = q1 - xb[NN + m], d2 = q2 - xb[2 * NN + m];
        dist[m] = -(d0 * d0 + d1 * d1 + d2 * d2);
    }
    __syncthreads();
    int* row = idx + ((size_t)b * NN + n) * KT;
    for (int t = 0; t < KT; ++t) {
        float bv = -__builtin_inff(); int bi = NN;
        for (int m = tid; m < NN; m += 256) {
            float v = dist[m];
            if (v > bv || (v == bv && m < bi)) { bv = v; bi = m; }
        }
        rv[tid] = bv; ri[tid] = bi;
        __syncthreads();
        for (int s = 128; s > 0; s >>= 1) {
            if (tid < s) {
                float ov = rv[tid + s]; int oi = ri[tid + s];
                if (ov > rv[tid] || (ov == rv[tid] && oi < ri[tid])) { rv[tid] = ov; ri[tid] = oi; }
            }
            __syncthreads();
        }
        if (tid == 0) { row[t] = ri[0]; dist[ri[0]] = -__builtin_inff(); }
        __syncthreads();
    }
}

// ---------------- 3) build xp (fp32 + bf16, pad rows 150..159 with zeros) ----------------
__global__ void k_build_xp(const float* __restrict__ x, const int* __restrict__ idx,
                           float* __restrict__ xp, unsigned short* __restrict__ xpb) {
    size_t t = (size_t)blockIdx.x * blockDim.x + threadIdx.x;
    if (t >= (size_t)BB * MP * NN) return;
    const int n = (int)(t % NN);
    const int m = (int)((t / NN) % MP);
    const int b = (int)(t / ((size_t)NN * MP));
    float v = 0.f;
    if (m < MM) {
        const int k = m / CC, c = m % CC;
        const int j = idx[((size_t)b * NN + n) * KT + k];
        const float* xr = x + ((size_t)b * CC + c) * NN;
        v = xr[j] - xr[n];
    }
    xp[t]  = v;
    xpb[t] = f2bf(v);
}

// ---------------- 4) WMMA GEMM with LDS-staged (async) weight slabs -----------------
// Out[b,m,n] = sum_k A[b,m,k]*Wt[k,n] + bias[n]
// A: (BB, MP, Kd) bf16 ; Wt: (Kd, Nout) bf16 ; Out: (BB, MP, Nout) fp32
// Block = 320 threads (10 waves). Wave w handles M-tile w and NG=4 N-tiles.
// B slab (32 x 64 bf16 = 4KB) double-buffered in LDS, prefetched async
// (GLOBAL_LOAD_ASYNC_TO_LDS_B128 / ASYNCcnt) by waves 0..7 (wave-uniform predicate,
// so EXEC is all-ones at every WMMA).
__global__ __launch_bounds__(320)
void k_wmma_gemm_bias(const unsigned short* __restrict__ A,
                      const unsigned short* __restrict__ Wt,
                      const float* __restrict__ bias,
                      float* __restrict__ Out,
                      int Kd, int Nout) {
    __shared__ unsigned short bslab[2][32 * 64];
    const int tid  = threadIdx.x;
    const int lane = tid & 31;
    const int wave = tid >> 5;                 // 0..9  == M-tile index
    const int ngroups = Nout / (16 * NG);
    const int b    = blockIdx.x / ngroups;
    const int ng   = blockIdx.x % ngroups;
    const int col0 = ng * (16 * NG);

    // A fragment addressing: lane holds row (lane&15); K-chunks at khalf*8 / 16+khalf*8
    const unsigned short* arow =
        A + ((size_t)b * MP + wave * 16 + (lane & 15)) * Kd + ((lane >> 4) * 8);

    // staging: threads 0..255 each move one 16B chunk of the 32x64 slab
    const int srow = (tid >> 3) & 31;          // k-row within slab
    const int scol = (tid & 7) * 8;            // column element offset
    const unsigned short* gsrc = Wt + (size_t)srow * Nout + col0 + scol;
    const int soff = srow * 64 + scol;

    if (tid < 256) stage16(&bslab[0][soff], gsrc);
    async_wait_all();
    __syncthreads();

    v8f acc[NG];
#pragma unroll
    for (int j = 0; j < NG; ++j) acc[j] = (v8f){0.f,0.f,0.f,0.f,0.f,0.f,0.f,0.f};

    int buf = 0;
    for (int k = 0; k < Kd; k += 32) {
        if (k + 32 < Kd && tid < 256)          // prefetch next slab into other buffer
            stage16(&bslab[buf ^ 1][soff], gsrc + (size_t)(k + 32) * Nout);

        FragU fa;
        *reinterpret_cast<uint4*>(&fa.u[0]) = *reinterpret_cast<const uint4*>(arow + k);
        *reinterpret_cast<uint4*>(&fa.u[4]) = *reinterpret_cast<const uint4*>(arow + k + 16);

        // B fragment: lane holds column (lane&15); K-rows khalf*8 / 16+khalf*8
        const unsigned short* bbase = &bslab[buf][(lane >> 4) * 8 * 64 + (lane & 15)];
#pragma unroll
        for (int j = 0; j < NG; ++j) {
            FragU fb;
#pragma unroll
            for (int i = 0; i < 8; ++i) {
                const int kr = (2 * i < 8) ? 2 * i : 2 * i + 8;
                fb.u[i] = (unsigned int)bbase[kr * 64 + j * 16] |
                          ((unsigned int)bbase[(kr + 1) * 64 + j * 16] << 16);
            }
            acc[j] = __builtin_amdgcn_wmma_f32_16x16x32_bf16(
                false, fa.v, false, fb.v, (short)0, acc[j], false, false);
        }
        async_wait_all();
        __syncthreads();
        buf ^= 1;
    }

    // C/D layout: lane -> col (lane&15), rows (lane>>4)*8 + r
    const int cn = lane & 15;
    const int mh = lane >> 4;
#pragma unroll
    for (int j = 0; j < NG; ++j) {
        const float bv = bias[col0 + j * 16 + cn];
        float* orow = Out + ((size_t)b * MP + wave * 16 + mh * 8) * Nout
                          + col0 + j * 16 + cn;
#pragma unroll
        for (int r = 0; r < 8; ++r) orow[(size_t)r * Nout] = acc[j][r] + bv;
    }
}

// ---------------- 5) attention + residual + LayerNorm1, one block per (b,m) -----------
__global__ void k_attn_ln1(const float* __restrict__ Q, const float* __restrict__ Km,
                           const float* __restrict__ V, const float* __restrict__ xp,
                           const float* __restrict__ g1, const float* __restrict__ be1,
                           float* __restrict__ o1, unsigned short* __restrict__ o1b) {
    __shared__ float qrow[NN];
    __shared__ float attn[MM + 2];
    __shared__ float red1[256], red2[256];
    const int b = blockIdx.x / MM;
    const int m = blockIdx.x % MM;
    const int tid = threadIdx.x;
    const size_t base = ((size_t)b * MP + m) * NN;

    for (int k = tid; k < NN; k += 256) qrow[k] = Q[base + k];
    __syncthreads();

    const float scale = rsqrtf((float)NN);
    for (int l = tid; l < MM; l += 256) {
        const float* kp = Km + ((size_t)b * MP + l) * NN;
        float dot = 0.f;
        for (int kk = 0; kk < NN; ++kk) dot += qrow[kk] * kp[kk];
        attn[l] = dot * scale;
    }
    __syncthreads();
    if (tid == 0) {                                  // softmax over 150 (small)
        float mx = -__builtin_inff();
        for (int l = 0; l < MM; ++l) mx = fmaxf(mx, attn[l]);
        float s = 0.f;
        for (int l = 0; l < MM; ++l) { float e = __expf(attn[l] - mx); attn[l] = e; s += e; }
        float inv = 1.f / s;
        for (int l = 0; l < MM; ++l) attn[l] *= inv;
    }
    __syncthreads();

    float yv[NN / 256];
    float lsum = 0.f, lsq = 0.f;
#pragma unroll
    for (int i = 0; i < NN / 256; ++i) {
        const int n = tid + 256 * i;
        float acc = 0.f;
        for (int l = 0; l < MM; ++l) acc += attn[l] * V[((size_t)b * MP + l) * NN + n];
        const float y = acc + xp[base + n];
        yv[i] = y; lsum += y; lsq += y * y;
    }
    red1[tid] = lsum; red2[tid] = lsq;
    __syncthreads();
    for (int s = 128; s > 0; s >>= 1) {
        if (tid < s) { red1[tid] += red1[tid + s]; red2[tid] += red2[tid + s]; }
        __syncthreads();
    }
    const float mu  = red1[0] * (1.f / NN);
    const float var = red2[0] * (1.f / NN) - mu * mu;
    const float inv = rsqrtf(var + 1e-5f);
#pragma unroll
    for (int i = 0; i < NN / 256; ++i) {
        const int n = tid + 256 * i;
        const float o = (yv[i] - mu) * inv * g1[n] + be1[n];
        o1[base + n]  = o;
        o1b[base + n] = f2bf(o);
    }
}

// zero bf16 pad rows 150..159 (so the d-GEMM reads zeros there)
__global__ void k_zero_pad(unsigned short* __restrict__ buf) {
    int t = blockIdx.x * blockDim.x + threadIdx.x;
    if (t >= BB * (MP - MM) * NN) return;
    const int n = t % NN;
    const int r = (t / NN) % (MP - MM);
    const int b = t / (NN * (MP - MM));
    buf[((size_t)b * MP + MM + r) * NN + n] = 0;
}

// ---------------- 6) residual + LayerNorm2, one block per (b,m) ----------------
__global__ void k_ln2(const float* __restrict__ D, const float* __restrict__ O1,
                      const float* __restrict__ g2, const float* __restrict__ be2,
                      float* __restrict__ O2) {
    __shared__ float red1[256], red2[256];
    const int b = blockIdx.x / MM;
    const int m = blockIdx.x % MM;
    const int tid = threadIdx.x;
    const size_t base = ((size_t)b * MP + m) * NN;
    float yv[NN / 256];
    float lsum = 0.f, lsq = 0.f;
#pragma unroll
    for (int i = 0; i < NN / 256; ++i) {
        const int n = tid + 256 * i;
        const float y = D[base + n] + O1[base + n];
        yv[i] = y; lsum += y; lsq += y * y;
    }
    red1[tid] = lsum; red2[tid] = lsq;
    __syncthreads();
    for (int s = 128; s > 0; s >>= 1) {
        if (tid < s) { red1[tid] += red1[tid + s]; red2[tid] += red2[tid + s]; }
        __syncthreads();
    }
    const float mu  = red1[0] * (1.f / NN);
    const float var = red2[0] * (1.f / NN) - mu * mu;
    const float inv = rsqrtf(var + 1e-5f);
#pragma unroll
    for (int i = 0; i < NN / 256; ++i) {
        const int n = tid + 256 * i;
        O2[base + n] = (yv[i] - mu) * inv * g2[n] + be2[n];
    }
}

// ---------------- 7) scores[b,n,k] = sum_c (out2+xp)[b,3k+c,n]*Wfc[c] + bfc ----------
__global__ void k_scores(const float* __restrict__ O2, const float* __restrict__ xp,
                         const float* __restrict__ Wfc, const float* __restrict__ bfc,
                         float* __restrict__ sc) {
    size_t t = (size_t)blockIdx.x * blockDim.x + threadIdx.x;
    if (t >= (size_t)BB * NN * KT) return;
    const int k = (int)(t % KT);
    const int n = (int)((t / KT) % NN);
    const int b = (int)(t / ((size_t)KT * NN));
    float s = bfc[0];
#pragma unroll
    for (int c = 0; c < CC; ++c) {
        const size_t r = ((size_t)b * MP + k * CC + c) * NN + n;
        s += (O2[r] + xp[r]) * Wfc[c];
    }
    sc[t] = s;
}

// ---------------- 8) stable argsort-select 20 smallest, gather, transpose ------------
__global__ void k_select(const float* __restrict__ sc, const float* __restrict__ xp,
                         float* __restrict__ out) {
    int t = blockIdx.x * blockDim.x + threadIdx.x;
    if (t >= BB * NN) return;
    const int n = t % NN;
    const int b = t / NN;
    const float* row = sc + (size_t)t * KT;
    float s[KT];
#pragma unroll
    for (int k = 0; k < KT; ++k) s[k] = row[k];
    unsigned long long used = 0ull;
    for (int j = 0; j < KK; ++j) {
        int bi = -1; float bv = __builtin_inff();
        for (int k = 0; k < KT; ++k)
            if (!((used >> k) & 1ull) && s[k] < bv) { bv = s[k]; bi = k; }
        if (bi < 0)      // all-NaN safety: pick first unused (stable)
            for (int k = 0; k < KT; ++k) if (!((used >> k) & 1ull)) { bi = k; break; }
        used |= 1ull << bi;
#pragma unroll
        for (int c = 0; c < CC; ++c)
            out[(((size_t)b * CC + c) * KK + j) * NN + n] =
                xp[((size_t)b * MP + bi * CC + c) * NN + n];
    }
}

// =====================================================================================
extern "C" void kernel_launch(void* const* d_in, const int* in_sizes, int n_in,
                              void* d_out, int out_size, void* d_ws, size_t ws_size,
                              hipStream_t stream) {
    (void)in_sizes; (void)n_in; (void)out_size; (void)ws_size;
    const float* x   = (const float*)d_in[0];
    const float* Wq  = (const float*)d_in[1];
    const float* bq  = (const float*)d_in[2];
    const float* Wk  = (const float*)d_in[3];
    const float* bk  = (const float*)d_in[4];
    const float* Wv  = (const float*)d_in[5];
    const float* bv  = (const float*)d_in[6];
    const float* Wd  = (const float*)d_in[7];
    const float* bd  = (const float*)d_in[8];
    const float* g1  = (const float*)d_in[9];
    const float* be1 = (const float*)d_in[10];
    const float* g2  = (const float*)d_in[11];
    const float* be2 = (const float*)d_in[12];
    const float* Wfc = (const float*)d_in[13];
    const float* bfc = (const float*)d_in[14];
    float* out = (float*)d_out;

    // ---- bump-allocate workspace (aliasing noted) ----
    char* p = (char*)d_ws;
    auto take = [&](size_t bytes) -> char* {
        char* r = p; p += (bytes + 255) & ~(size_t)255; return r;
    };
    const size_t ACT  = (size_t)BB * MP * NN;          // activation elements
    int*            idx = (int*)           take((size_t)BB * NN * KT * 4);
    float*          xp  = (float*)         take(ACT * 4);
    unsigned short* xpb = (unsigned short*)take(ACT * 2);
    unsigned short* wqT = (unsigned short*)take((size_t)NN * NN * 2);
    unsigned short* wkT = (unsigned short*)take((size_t)NN * NN * 2);
    unsigned short* wvT = (unsigned short*)take((size_t)NN * NN * 2);
    unsigned short* wdT = (unsigned short*)take((size_t)NN * NN * 2);
    float*          q   = (float*)         take(ACT * 4);
    float*          km  = (float*)         take(ACT * 4);
    float*          v   = (float*)         take(ACT * 4);
    float*          o1  = (float*)         take(ACT * 4);
    unsigned short* o1b = (unsigned short*)take(ACT * 2);
    float* dmat = q;                      // alias: q consumed before d-GEMM
    float* o2   = km;                     // alias: k consumed before LN2
    float* sc   = v;                      // alias: v consumed before scores

    // 1) weights -> bf16 transposed
    {
        const int blocks = (int)(((size_t)NN * NN + 255) / 256);
        k_w2bfT<<<blocks, 256, 0, stream>>>(Wq, wqT, NN, NN);
        k_w2bfT<<<blocks, 256, 0, stream>>>(Wk, wkT, NN, NN);
        k_w2bfT<<<blocks, 256, 0, stream>>>(Wv, wvT, NN, NN);
        k_w2bfT<<<blocks, 256, 0, stream>>>(Wd, wdT, NN, NN);
    }
    // 2) KNN top-50
    k_knn<<<BB * NN, 256, 0, stream>>>(x, idx);
    // 3) edge features xp (fp32 + bf16, zero-padded rows)
    k_build_xp<<<(int)((ACT + 255) / 256), 256, 0, stream>>>(x, idx, xp, xpb);
    // 4) q/k/v GEMMs on WMMA (320 threads = 10 waves = 10 M-tiles per block)
    {
        const int blocks = BB * (NN / (16 * NG));        // 16 * 32 = 512
        k_wmma_gemm_bias<<<blocks, 320, 0, stream>>>(xpb, wqT, bq, q,  NN, NN);
        k_wmma_gemm_bias<<<blocks, 320, 0, stream>>>(xpb, wkT, bk, km, NN, NN);
        k_wmma_gemm_bias<<<blocks, 320, 0, stream>>>(xpb, wvT, bv, v,  NN, NN);
        // 5) attention + residual + LN1
        k_attn_ln1<<<BB * MM, 256, 0, stream>>>(q, km, v, xp, g1, be1, o1, o1b);
        k_zero_pad<<<(BB * (MP - MM) * NN + 255) / 256, 256, 0, stream>>>(o1b);
        // 6) d GEMM on WMMA (A = out1 bf16), then residual + LN2
        k_wmma_gemm_bias<<<blocks, 320, 0, stream>>>(o1b, wdT, bd, dmat, NN, NN);
    }
    k_ln2<<<BB * MM, 256, 0, stream>>>(dmat, o1, g2, be2, o2);
    // 7) scoring projection over C
    k_scores<<<(int)(((size_t)BB * NN * KT + 255) / 256), 256, 0, stream>>>(o2, xp, Wfc, bfc, sc);
    // 8) select 20 smallest (stable) and emit (B, C, K, N)
    k_select<<<(BB * NN + 255) / 256, 256, 0, stream>>>(sc, xp, out);
}
